// GraphEncoder_266287973074
// MI455X (gfx1250) — matrix operand order repaired and checked by reading, not compiled, not run on hardware
//
#include <hip/hip_runtime.h>
#include <hip/hip_bf16.h>
#include <stdint.h>

#define NN  50000
#define NE  1600000
#define DIM 128

typedef __bf16 bf16_t;
typedef __attribute__((ext_vector_type(16))) __bf16 v16bf;
typedef __attribute__((ext_vector_type(8)))  __bf16 v8bf;
typedef __attribute__((ext_vector_type(8)))  float  v8f;

// Non-returning, relaxed, device-scope f32 atomic add -> global_atomic_add_f32
__device__ __forceinline__ void atomic_add_relaxed(float* p, float v) {
  (void)__hip_atomic_fetch_add(p, v, __ATOMIC_RELAXED, __HIP_MEMORY_SCOPE_AGENT);
}

// ---------------- prep kernels (once per launch) ----------------

__global__ void k_init_deg(float* deg) {
  int i = blockIdx.x * blockDim.x + threadIdx.x;
  if (i < NN) deg[i] = 1.0f;                       // self-loop
}

__global__ void k_deg_accum(const int* __restrict__ ei, float* __restrict__ deg) {
  int e = blockIdx.x * blockDim.x + threadIdx.x;
  if (e < NE) atomic_add_relaxed(&deg[ei[NE + e]], 1.0f);   // deg over col
}

__global__ void k_dinv(float* deg) {
  int i = blockIdx.x * blockDim.x + threadIdx.x;
  if (i < NN) deg[i] = rsqrtf(deg[i]);             // deg >= 1 always
}

__global__ void k_norm(const int* __restrict__ ei, const float* __restrict__ dinv,
                       float* __restrict__ norm) {
  int e = blockIdx.x * blockDim.x + threadIdx.x;
  if (e < NE) norm[e] = dinv[ei[e]] * dinv[ei[NE + e]];
}

__global__ void k_x_to_bf16(const float* __restrict__ x, bf16_t* __restrict__ xb) {
  int i = blockIdx.x * blockDim.x + threadIdx.x;
  if (i < NN * DIM) xb[i] = (bf16_t)x[i];
}

// Wt_bf16[l][n][k] = W_l[k][n]  (transpose so B fragments are contiguous in K)
__global__ void k_prep_w(const float* __restrict__ w0, const float* __restrict__ w1,
                         const float* __restrict__ w2, bf16_t* __restrict__ wbt) {
  int t = blockIdx.x * blockDim.x + threadIdx.x;
  if (t >= 3 * DIM * DIM) return;
  int l = t >> 14, idx = t & (DIM * DIM - 1);
  int n = idx >> 7, k = idx & 127;
  const float* w = (l == 0) ? w0 : (l == 1) ? w1 : w2;
  wbt[t] = (bf16_t)w[k * DIM + n];
}

__global__ void k_zero(float* acc) {
  int i = blockIdx.x * blockDim.x + threadIdx.x;
  if (i < NN * DIM) acc[i] = 0.0f;
}

// ---------------- dense transform: h = x @ W  (bf16 WMMA, f32 acc) -------
// One wave per 16-row stripe; wave computes all 8 column tiles.
// A frag (16x32 bf16, ISA layout): lane(0-15)=row, K = kc + s*8 + [0..7] in v0..3,
//                                  K = kc + 16 + s*8 + [0..7] in v4..7 (s = lane>>4).
// B frag (32x16 bf16): lane = col (mod 16), K same as A frag, pairs per VGPR.
// C/D (16x16 f32): c[v] holds row v + s*8, col = lane&15.
__global__ __launch_bounds__(256) void k_gemm(const bf16_t* __restrict__ xb,
                                              const bf16_t* __restrict__ wt,
                                              float* __restrict__ h) {
  const int wave   = threadIdx.x >> 5;
  const int lane   = threadIdx.x & 31;
  const int stripe = blockIdx.x * 8 + wave;      // 16-row stripe
  if (stripe * 16 >= NN) return;                 // wave-uniform: EXEC stays full
  const int r = lane & 15;
  const int s = lane >> 4;
  const int row0 = stripe * 16;

  v8f c[8];
#pragma unroll
  for (int nt = 0; nt < 8; ++nt)
#pragma unroll
    for (int v = 0; v < 8; ++v) c[nt][v] = 0.0f;

  const bf16_t* arow = xb + (size_t)(row0 + r) * DIM;
#pragma unroll
  for (int kc = 0; kc < DIM; kc += 32) {
    v8bf lo = *(const v8bf*)(arow + kc + s * 8);
    v8bf hi = *(const v8bf*)(arow + kc + 16 + s * 8);
    v16bf a;
#pragma unroll
    for (int i = 0; i < 8; ++i) { a[i] = lo[i]; a[8 + i] = hi[i]; }

#pragma unroll
    for (int nt = 0; nt < 8; ++nt) {
      const bf16_t* q = wt + (size_t)(nt * 16 + r) * DIM + kc + s * 8;
      v8bf b0 = *(const v8bf*)(q);
      v8bf b1 = *(const v8bf*)(q + 16);
      v16bf b;
#pragma unroll
      for (int i = 0; i < 8; ++i) { b[i] = b0[i]; b[8 + i] = b1[i]; }
      c[nt] = __builtin_amdgcn_wmma_f32_16x16x32_bf16(
          false, a, false, b, (short)0, c[nt], false, false);
    }
  }

#pragma unroll
  for (int nt = 0; nt < 8; ++nt)
#pragma unroll
    for (int v = 0; v < 8; ++v)
      h[(size_t)(row0 + v + s * 8) * DIM + nt * 16 + r] = c[nt][v];
}

// ---------------- edge aggregation: acc[col] += h[row] * norm -------------
// One wave per edge: 512B coalesced row gather + 128 f32 hardware atomics
// (4 per lane, contiguous) resolved at the L2 atomic units. Whole per-layer
// working set (~70 MB) is L2-resident on MI455X's 192 MB L2.
__global__ void k_scatter(const float* __restrict__ h, const int* __restrict__ ei,
                          const float* __restrict__ norm, float* __restrict__ acc) {
  int tid = blockIdx.x * blockDim.x + threadIdx.x;   // NE*32 = 51.2M < 2^31
  int e = tid >> 5;
  int c = (tid & 31) * 4;
  if (e >= NE) return;
  int src = ei[e];
  int dst = ei[NE + e];
  float w = norm[e];
  float4 v = *(const float4*)(h + (size_t)src * DIM + c);
  float* p = acc + (size_t)dst * DIM + c;
  atomic_add_relaxed(p + 0, v.x * w);
  atomic_add_relaxed(p + 1, v.y * w);
  atomic_add_relaxed(p + 2, v.z * w);
  atomic_add_relaxed(p + 3, v.w * w);
}

// ---------------- epilogue: self-loop + bias + ReLU, recast to bf16 -------
__global__ void k_epilogue(const float* __restrict__ acc, const float* __restrict__ h,
                           const float* __restrict__ dinv, const float* __restrict__ bias,
                           bf16_t* __restrict__ xb_next, float* __restrict__ fout) {
  int tid = blockIdx.x * blockDim.x + threadIdx.x;
  if (tid >= NN * DIM) return;
  int i = tid >> 7, f = tid & 127;
  float d = dinv[i];
  float val = acc[tid] + h[tid] * d * d + bias[f];
  val = fmaxf(val, 0.0f);
  xb_next[tid] = (bf16_t)val;
  if (fout) fout[tid] = val;
}

// --------------------------------------------------------------------------

static inline size_t align256(size_t x) { return (x + 255) & ~(size_t)255; }

extern "C" void kernel_launch(void* const* d_in, const int* in_sizes, int n_in,
                              void* d_out, int out_size, void* d_ws, size_t ws_size,
                              hipStream_t stream) {
  (void)in_sizes; (void)n_in; (void)out_size; (void)ws_size;
  const float* x  = (const float*)d_in[0];
  const int*   ei = (const int*)d_in[1];
  const float* W[3] = {(const float*)d_in[2], (const float*)d_in[4], (const float*)d_in[6]};
  const float* B[3] = {(const float*)d_in[3], (const float*)d_in[5], (const float*)d_in[7]};

  char* p = (char*)d_ws;
  float*  dinv = (float*)p;  p += align256((size_t)NN * 4);
  float*  norm = (float*)p;  p += align256((size_t)NE * 4);
  bf16_t* wbt  = (bf16_t*)p; p += align256((size_t)3 * DIM * DIM * 2);
  bf16_t* xb   = (bf16_t*)p; p += align256((size_t)NN * DIM * 2);
  float*  h    = (float*)p;  p += align256((size_t)NN * DIM * 4);
  float*  acc  = (float*)p;  p += align256((size_t)NN * DIM * 4);

  const int TB = 256;
  // --- graph normalization precompute ---
  k_init_deg <<<(NN + TB - 1) / TB, TB, 0, stream>>>(dinv);
  k_deg_accum<<<(NE + TB - 1) / TB, TB, 0, stream>>>(ei, dinv);
  k_dinv     <<<(NN + TB - 1) / TB, TB, 0, stream>>>(dinv);
  k_norm     <<<(NE + TB - 1) / TB, TB, 0, stream>>>(ei, dinv, norm);
  // --- precision prep ---
  k_x_to_bf16<<<(NN * DIM + TB - 1) / TB, TB, 0, stream>>>(x, xb);
  k_prep_w   <<<(3 * DIM * DIM + TB - 1) / TB, TB, 0, stream>>>(W[0], W[1], W[2], wbt);

  const int stripes    = NN / 16;                       // 3125 exactly
  const int gemm_grid  = (stripes + 7) / 8;             // 8 waves / block
  const int scat_grid  = (NE * 32 + TB - 1) / TB;       // 200000 blocks
  const int nd_grid    = (NN * DIM + TB - 1) / TB;

  for (int l = 0; l < 3; ++l) {
    k_gemm    <<<gemm_grid, TB, 0, stream>>>(xb, wbt + (size_t)l * DIM * DIM, h);
    k_zero    <<<nd_grid, TB, 0, stream>>>(acc);
    k_scatter <<<scat_grid, TB, 0, stream>>>(h, ei, norm, acc);
    k_epilogue<<<nd_grid, TB, 0, stream>>>(acc, h, dinv, B[l], xb,
                                           (l == 2) ? (float*)d_out : nullptr);
  }
}
